// Attention_79688823210649
// MI455X (gfx1250) — matrix operand order repaired and checked by reading, not compiled, not run on hardware
//
#include <hip/hip_runtime.h>

// ---------------------------------------------------------------- types
typedef __attribute__((ext_vector_type(16))) __bf16 v16bf;
typedef __attribute__((ext_vector_type(8)))  __bf16 v8bf;
typedef __attribute__((ext_vector_type(8)))  float  v8f;

#define S_LEN 2048
#define D_DIM 128
#define H_NUM 8
#define B_NUM 2
#define NEGV  (-1000000000.0f)
#define QSCALE 0.08838834764831845f   // 1/sqrt(128)

// ------------------------------------------------------------ WMMA core
__device__ __forceinline__ v8f wmma_bf16(v16bf a, v16bf b, v8f c) {
  return __builtin_amdgcn_wmma_f32_16x16x32_bf16(
      false, a, false, b, (short)0, c, false, false);
}

__device__ __forceinline__ v16bf cat8(v8bf lo, v8bf hi) {
  return __builtin_shufflevector(lo, hi, 0, 1, 2, 3, 4, 5, 6, 7,
                                 8, 9, 10, 11, 12, 13, 14, 15);
}

// A fragment from row-major [M][K] memory; rowptr -> row (lane&15), k-tile base.
// lanes 0-15: K={0..7,16..23}; lanes 16-31: K={8..15,24..31}  (ISA 7.12.2)
__device__ __forceinline__ v16bf frag_a(const __bf16* rowptr, int half) {
  v8bf lo = *(const v8bf*)(rowptr + (half << 3));
  v8bf hi = *(const v8bf*)(rowptr + 16 + (half << 3));
  return cat8(lo, hi);
}

// B fragment from [N][K] (K-contiguous) memory: same K order as frag_a.
__device__ __forceinline__ v16bf frag_b(const __bf16* colptr, int half) {
  v8bf lo = *(const v8bf*)(colptr + (half << 3));
  v8bf hi = *(const v8bf*)(colptr + 16 + (half << 3));
  return cat8(lo, hi);
}

// ------------------------------------------------------ wave reductions
__device__ __forceinline__ float wred_max(float v) {
#pragma unroll
  for (int o = 16; o > 0; o >>= 1) v = fmaxf(v, __shfl_xor(v, o, 32));
  return v;
}
__device__ __forceinline__ float wred_sum(float v) {
#pragma unroll
  for (int o = 16; o > 0; o >>= 1) v += __shfl_xor(v, o, 32);
  return v;
}

// --------------------------------------------------------- pack kernels
__global__ void cast_f32_to_bf16(const float* __restrict__ in,
                                 __bf16* __restrict__ out, int n) {
  int i = blockIdx.x * 256 + threadIdx.x;
  if (i < n) out[i] = (__bf16)in[i];
}

// w_qk [O=2048][I=128][T=5] -> wqkT [O][K=640], K = t*128+i (matches im2col A)
__global__ void pack_wqkT(const float* __restrict__ w, __bf16* __restrict__ out) {
  int idx = blockIdx.x * 256 + threadIdx.x;
  if (idx < 2048 * 640) {
    int o = idx / 640;
    int rem = idx - o * 640;
    int t = rem >> 7, i = rem & 127;
    out[idx] = (__bf16)w[(o * 128 + i) * 5 + t];
  }
}
// w_v [K=128][N=1024] -> wvT [N][K]
__global__ void pack_wvT(const float* __restrict__ w, __bf16* __restrict__ out) {
  int idx = blockIdx.x * 256 + threadIdx.x;
  if (idx < 1024 * 128) {
    int n = idx >> 7, k = idx & 127;
    out[idx] = (__bf16)w[k * 1024 + n];
  }
}
// w_proj [K=1024][N=128] -> wpT [N][K]
__global__ void pack_wpT(const float* __restrict__ w, __bf16* __restrict__ out) {
  int idx = blockIdx.x * 256 + threadIdx.x;
  if (idx < 128 * 1024) {
    int n = idx >> 10, k = idx & 1023;
    out[idx] = (__bf16)w[k * 128 + n];
  }
}

// ----------------------------------------------------------- QK conv-GEMM
// M=4096 (b,s), K=640, N=2048. Block 32x256 (8 waves, each 16x64, 4 accums).
__global__ __launch_bounds__(256) void qk_gemm(
    const __bf16* __restrict__ xb, const __bf16* __restrict__ wqkT,
    const float* __restrict__ b_qk, __bf16* __restrict__ qb,
    __bf16* __restrict__ kb) {
  int tid = threadIdx.x;
  int lane = tid & 31, wave = tid >> 5;
  int wm = wave >> 2, wn = wave & 3;
  int half = lane >> 4, nn = lane & 15;
  int gm0 = blockIdx.y * 32 + wm * 16;
  int gn0 = blockIdx.x * 256 + wn * 64;
  int gm = gm0 + nn;                    // A row handled by this lane
  int b = gm >> 11, s = gm & 2047;
  v8f acc[4] = {{}, {}, {}, {}};
  for (int kt = 0; kt < 640; kt += 32) {
    int t = kt >> 7, i0 = kt & 127;     // K-chunk lies inside one conv tap
    int ss = s - 4 + t;
    v16bf a = {};
    if (ss >= 0) {
      const __bf16* ap = xb + ((size_t)b * S_LEN + ss) * D_DIM + i0;
      a = frag_a(ap, half);
    }
#pragma unroll
    for (int u = 0; u < 4; ++u) {
      const __bf16* bp = wqkT + (size_t)(gn0 + u * 16 + nn) * 640 + kt;
      acc[u] = wmma_bf16(a, frag_b(bp, half), acc[u]);
    }
  }
#pragma unroll
  for (int u = 0; u < 4; ++u) {
    int col = gn0 + u * 16 + nn;
    float bias = b_qk[col];
#pragma unroll
    for (int r = 0; r < 8; ++r) {
      int gm2 = gm0 + 8 * half + r;
      int b2 = gm2 >> 11, s2 = gm2 & 2047;
      float val = acc[u][r] + bias;
      if (col < 1024) {                 // query, fold 1/sqrt(D)
        int h = col >> 7, d = col & 127;
        qb[((size_t)(b2 * H_NUM + h) * S_LEN + s2) * D_DIM + d] =
            (__bf16)(val * QSCALE);
      } else {                          // key
        int c2 = col - 1024;
        int h = c2 >> 7, d = c2 & 127;
        kb[((size_t)(b2 * H_NUM + h) * S_LEN + s2) * D_DIM + d] = (__bf16)val;
      }
    }
  }
}

// ----------------------------------------------------------- value GEMM
// value = x @ w_v + b_v, stored TRANSPOSED: vbT[bh][d][s]. M=4096,K=128,N=1024.
// In this layout the 8 C rows of a lane are contiguous in s -> one v8bf store.
__global__ __launch_bounds__(256) void v_gemm(
    const __bf16* __restrict__ xb, const __bf16* __restrict__ wvT,
    const float* __restrict__ b_v, __bf16* __restrict__ vbT) {
  int tid = threadIdx.x;
  int lane = tid & 31, wave = tid >> 5;
  int wm = wave >> 2, wn = wave & 3;
  int half = lane >> 4, nn = lane & 15;
  int gm0 = blockIdx.y * 32 + wm * 16;
  int gn0 = blockIdx.x * 256 + wn * 64;
  const __bf16* ap0 = xb + (size_t)(gm0 + nn) * D_DIM;
  v8f acc[4] = {{}, {}, {}, {}};
  for (int kt = 0; kt < 128; kt += 32) {
    v16bf a = frag_a(ap0 + kt, half);
#pragma unroll
    for (int u = 0; u < 4; ++u) {
      const __bf16* bp = wvT + (size_t)(gn0 + u * 16 + nn) * 128 + kt;
      acc[u] = wmma_bf16(a, frag_b(bp, half), acc[u]);
    }
  }
  int gm2 = gm0 + 8 * half;             // first of 8 contiguous s rows
  int b2 = gm2 >> 11, s2 = gm2 & 2047;
#pragma unroll
  for (int u = 0; u < 4; ++u) {
    int col = gn0 + u * 16 + nn;
    float bias = b_v[col];
    int h = col >> 7, d = col & 127;
    v8bf pack;
#pragma unroll
    for (int r = 0; r < 8; ++r) pack[r] = (__bf16)(acc[u][r] + bias);
    *(v8bf*)&vbT[((size_t)(b2 * H_NUM + h) * D_DIM + d) * S_LEN + s2] = pack;
  }
}

// -------------------------------------------------- fused attention tile
// One block = one (b,h) x 16 query rows. Score strip in LDS (128KB of the
// 320KB WGP LDS). entmax-1.5 via register-resident bisection. Phase 2 writes
// the attention weights back bf16-packed IN PLACE, so phase 3 A-fragments are
// contiguous bf16; phase 3 B-fragments come straight from transposed v.
__global__ __launch_bounds__(256) void attn(
    const __bf16* __restrict__ qb, const __bf16* __restrict__ kb,
    const __bf16* __restrict__ vbT, const float* __restrict__ mask,
    __bf16* __restrict__ ctx) {
  __shared__ float sc[16][S_LEN];       // 131072 B
  int tid = threadIdx.x;
  int lane = tid & 31, wave = tid >> 5;
  int half = lane >> 4, nn = lane & 15;
  int bh = blockIdx.y;                  // b*8+h
  int q0 = blockIdx.x * 16;
  const __bf16* qbase = qb + (size_t)bh * S_LEN * D_DIM;
  const __bf16* kbase = kb + (size_t)bh * S_LEN * D_DIM;
  const __bf16* vtb   = vbT + (size_t)bh * D_DIM * S_LEN;

  // preload the 4 A fragments of the q tile (row m = lane&15)
  v16bf aq[4];
#pragma unroll
  for (int dc = 0; dc < 4; ++dc)
    aq[dc] = frag_a(qbase + (size_t)(q0 + nn) * D_DIM + dc * 32, half);

  // ---- phase 1: scores = (q/sqrt(D)) . k^T, masked, into LDS strip
  for (int jt = 0; jt < 16; ++jt) {
    int key0 = (jt * 8 + wave) * 16;
    v8f c = {};
#pragma unroll
    for (int dc = 0; dc < 4; ++dc) {
      // B[k=d][n=key]: two runs of 8 contiguous bf16 along d -> two 16B loads
      const __bf16* bp = kbase + (size_t)(key0 + nn) * D_DIM + dc * 32;
      c = wmma_bf16(aq[dc], frag_b(bp, half), c);
    }
#pragma unroll
    for (int r = 0; r < 8; ++r) {
      int row = 8 * half + r;
      float mv = mask[(size_t)(q0 + row) * S_LEN + key0 + nn];
      sc[row][key0 + nn] = c[r] * mv + NEGV * (1.0f - mv);
    }
  }
  __syncthreads();

  // ---- phase 2: entmax-1.5 per row (2 rows/wave); bisection for tau;
  //      write result back bf16-packed in place.
#pragma unroll
  for (int rr = 0; rr < 2; ++rr) {
    int row = wave * 2 + rr;
    float xr[64];
#pragma unroll
    for (int j = 0; j < 64; ++j) xr[j] = sc[row][lane + j * 32];
    float mx = -3.4e38f;
#pragma unroll
    for (int j = 0; j < 64; ++j) mx = fmaxf(mx, xr[j]);
    mx = wred_max(mx);
#pragma unroll
    for (int j = 0; j < 64; ++j) xr[j] = (xr[j] - mx) * 0.5f;
    // root of f(tau)=sum((x-tau)_+^2)-1 on [-1,0]; f(-1)>=1, f(0)=0
    float lo = -1.0f, hi = 0.0f;
    for (int it = 0; it < 30; ++it) {
      float mid = 0.5f * (lo + hi);
      float ssum = 0.0f;
#pragma unroll
      for (int j = 0; j < 64; ++j) {
        float t = fmaxf(xr[j] - mid, 0.0f);
        ssum += t * t;
      }
      ssum = wred_sum(ssum);
      if (ssum >= 1.0f) lo = mid; else hi = mid;
    }
    float tau = 0.5f * (lo + hi);
    __bf16* brow = (__bf16*)&sc[row][0];   // in-place bf16 repack
#pragma unroll
    for (int j = 0; j < 64; ++j) {
      float t = fmaxf(xr[j] - tau, 0.0f);
      brow[lane + j * 32] = (__bf16)(t * t);
    }
  }
  __syncthreads();

  // ---- phase 3: out(16x128) = att(16x2048) @ v(2048x128); wave -> 16 cols.
  // Two interleaved accumulator chains to hide XDL latency.
  int d0 = wave * 16;
  v8f cacc0 = {}, cacc1 = {};
  const __bf16* arow = (const __bf16*)&sc[nn][0];
  const __bf16* bcol = vtb + (size_t)(d0 + nn) * S_LEN;
  for (int k0 = 0; k0 < S_LEN; k0 += 64) {
    if (k0 + 64 < S_LEN) __builtin_prefetch(bcol + k0 + 64, 0, 0);
    v16bf a0 = frag_a(arow + k0, half);
    v16bf a1 = frag_a(arow + k0 + 32, half);
    cacc0 = wmma_bf16(a0, frag_b(bcol + k0, half), cacc0);
    cacc1 = wmma_bf16(a1, frag_b(bcol + k0 + 32, half), cacc1);
  }
  v8f cacc = cacc0 + cacc1;
  int bb = bh >> 3, h = bh & 7;
#pragma unroll
  for (int r = 0; r < 8; ++r) {
    int row = 8 * half + r;
    ctx[((size_t)bb * S_LEN + q0 + row) * 1024 + h * D_DIM + d0 + nn] =
        (__bf16)cacc[r];
  }
}

// ------------------------------------------------------------ proj GEMM
// out = ctx @ w_proj + b_proj. M=4096, K=1024, N=128. f32 output.
// Block 64x128 (8 waves: 4 m-tiles x 2 n-groups, each wave 16x64).
__global__ __launch_bounds__(256) void proj_gemm(
    const __bf16* __restrict__ ctx, const __bf16* __restrict__ wpT,
    const float* __restrict__ b_proj, float* __restrict__ out) {
  int tid = threadIdx.x;
  int lane = tid & 31, wave = tid >> 5;
  int wm = wave >> 1, wn = wave & 1;
  int half = lane >> 4, nn = lane & 15;
  int gm0 = blockIdx.y * 64 + wm * 16;
  int gn0 = wn * 64;
  const __bf16* ap0 = ctx + (size_t)(gm0 + nn) * 1024;
  v8f acc[4] = {{}, {}, {}, {}};
  for (int kt = 0; kt < 1024; kt += 32) {
    v16bf a = frag_a(ap0 + kt, half);
#pragma unroll
    for (int u = 0; u < 4; ++u) {
      const __bf16* bp = wpT + (size_t)(gn0 + u * 16 + nn) * 1024 + kt;
      acc[u] = wmma_bf16(a, frag_b(bp, half), acc[u]);
    }
  }
#pragma unroll
  for (int u = 0; u < 4; ++u) {
    int col = gn0 + u * 16 + nn;
    float bias = b_proj[col];
#pragma unroll
    for (int r = 0; r < 8; ++r) {
      int gm2 = gm0 + 8 * half + r;
      out[(size_t)gm2 * 128 + col] = acc[u][r] + bias;
    }
  }
}

// --------------------------------------------------------------- launch
extern "C" void kernel_launch(void* const* d_in, const int* in_sizes, int n_in,
                              void* d_out, int out_size, void* d_ws,
                              size_t ws_size, hipStream_t stream) {
  (void)in_sizes; (void)n_in; (void)out_size; (void)ws_size;
  const float* x      = (const float*)d_in[0];
  const float* mask   = (const float*)d_in[1];
  const float* w_qk   = (const float*)d_in[2];
  const float* b_qk   = (const float*)d_in[3];
  const float* w_v    = (const float*)d_in[4];
  const float* b_v    = (const float*)d_in[5];
  const float* w_proj = (const float*)d_in[6];
  const float* b_proj = (const float*)d_in[7];
  float* out = (float*)d_out;

  char* p = (char*)d_ws;
  __bf16* xb   = (__bf16*)p; p += (size_t)B_NUM * S_LEN * D_DIM * 2;
  __bf16* wqkT = (__bf16*)p; p += (size_t)2048 * 640 * 2;
  __bf16* wvT  = (__bf16*)p; p += (size_t)1024 * 128 * 2;
  __bf16* wpT  = (__bf16*)p; p += (size_t)128 * 1024 * 2;
  __bf16* qb   = (__bf16*)p; p += (size_t)B_NUM * H_NUM * S_LEN * D_DIM * 2;
  __bf16* kb   = (__bf16*)p; p += (size_t)B_NUM * H_NUM * S_LEN * D_DIM * 2;
  __bf16* vbT  = (__bf16*)p; p += (size_t)B_NUM * H_NUM * S_LEN * D_DIM * 2;
  __bf16* ctx  = (__bf16*)p; p += (size_t)B_NUM * S_LEN * 1024 * 2;

  const int nx = B_NUM * S_LEN * D_DIM;  // 524288
  cast_f32_to_bf16<<<(nx + 255) / 256, 256, 0, stream>>>(x, xb, nx);
  pack_wqkT<<<(2048 * 640 + 255) / 256, 256, 0, stream>>>(w_qk, wqkT);
  pack_wvT <<<(1024 * 128 + 255) / 256, 256, 0, stream>>>(w_v, wvT);
  pack_wpT <<<(128 * 1024 + 255) / 256, 256, 0, stream>>>(w_proj, wpT);

  qk_gemm  <<<dim3(8, 128),  256, 0, stream>>>(xb, wqkT, b_qk, qb, kb);
  v_gemm   <<<dim3(4, 128),  256, 0, stream>>>(xb, wvT, b_v, vbT);
  attn     <<<dim3(128, 16), 256, 0, stream>>>(qb, kb, vbT, mask, ctx);
  proj_gemm<<<dim3(1, 64),   256, 0, stream>>>(ctx, wpT, b_proj, out);
}
